// NonLocalMask_15848429322565
// MI455X (gfx1250) — matrix operand order repaired and checked by reading, not compiled, not run on hardware
//
#include <hip/hip_runtime.h>

typedef __attribute__((ext_vector_type(16))) __bf16 bf16x16;
typedef __attribute__((ext_vector_type(8)))  float  f32x8;

#define IC 256
#define NPOS 4096
#define HW 16384

#if defined(__has_builtin)
#if __has_builtin(__builtin_amdgcn_global_load_async_to_lds_b128) && \
    __has_builtin(__builtin_amdgcn_s_wait_asynccnt)
#define HAVE_ASYNC_LDS 1
#endif
#endif

#ifdef HAVE_ASYNC_LDS
// Builtin signature (from hipcc diagnostics): arg0 = v4i in addrspace(1) (global),
// arg1 = v4i in addrspace(3) (LDS), then two immediates.
typedef int v4i_vs __attribute__((vector_size(16)));
typedef __attribute__((address_space(1))) v4i_vs* as1_v4i;
typedef __attribute__((address_space(3))) v4i_vs* as3_v4i;
#endif

__device__ __forceinline__ unsigned short f2bf(float f) {
  union { float f; unsigned u; } v; v.f = f;
  return (unsigned short)((v.u + 0x7FFFu + ((v.u >> 16) & 1u)) >> 16);
}

union FragU { bf16x16 v; uint4 q[2]; };

__device__ __forceinline__ f32x8 zero8() {
  f32x8 z = {0.f,0.f,0.f,0.f,0.f,0.f,0.f,0.f};
  return z;
}

// A fragment 16x32 bf16 from row-major [rows][ld]; rows rbase..rbase+15, K = k0..k0+31
// lane (g = lane>>4, m = lane&15): v[0..7] = row m, K = k0+8g..; v[8..15] = row m, K = k0+16+8g..
__device__ __forceinline__ bf16x16 load_a_frag(const unsigned short* base, int ld, int rbase, int k0) {
  int lane = threadIdx.x & 31;
  int g = lane >> 4, m = lane & 15;
  const unsigned short* p = base + (size_t)(rbase + m) * ld + k0 + 8 * g;
  FragU u;
  u.q[0] = *(const uint4*)p;
  u.q[1] = *(const uint4*)(p + 16);
  return u.v;
}

// B fragment 32x16 bf16 supplied as B^T row-major [N][ld]; output cols nbase..nbase+15, K = k0..k0+31
// lane (g, n): v[0..7] = B^T[n][k0+8g..]; v[8..15] = B^T[n][k0+16+8g..]
__device__ __forceinline__ bf16x16 load_b_frag(const unsigned short* base, int ld, int nbase, int k0) {
  int lane = threadIdx.x & 31;
  int g = lane >> 4, n = lane & 15;
  const unsigned short* p = base + (size_t)(nbase + n) * ld + k0 + 8 * g;
  FragU u;
  u.q[0] = *(const uint4*)p;
  u.q[1] = *(const uint4*)(p + 16);
  return u.v;
}

__device__ __forceinline__ f32x8 wmma_bf(bf16x16 a, bf16x16 b, f32x8 c) {
  return __builtin_amdgcn_wmma_f32_16x16x32_bf16(false, a, false, b, (short)0, c, false, false);
}

// ---------------- 1) weights -> bf16 ----------------
__global__ void k_w2bf(const float* tw, const float* pw, const float* gw, unsigned short* Wbf) {
  int idx = blockIdx.x * 256 + threadIdx.x;
  if (idx >= 3 * 65536) return;
  int which = idx >> 16, e = idx & 65535;
  const float* s = (which == 0) ? tw : (which == 1) ? pw : gw;
  Wbf[idx] = f2bf(s[e]);
}

// ---------------- 2) x [b,256,4096] f32 -> XT [b,4096,256] bf16 ----------------
__global__ void k_xT(const float* X, unsigned short* XT) {
  __shared__ unsigned short t[64][65];
  int b = blockIdx.z;
  const float* s = X + (size_t)b * IC * NPOS;
  unsigned short* d = XT + (size_t)b * NPOS * IC;
  int n0 = blockIdx.x * 64, i0 = blockIdx.y * 64;
#pragma unroll
  for (int k = 0; k < 16; ++k) {
    int idx = threadIdx.x + k * 256;
    int r = idx >> 6, c = idx & 63;
    t[c][r] = f2bf(s[(size_t)(i0 + r) * NPOS + n0 + c]);
  }
  __syncthreads();
#pragma unroll
  for (int k = 0; k < 16; ++k) {
    int idx = threadIdx.x + k * 256;
    int r = idx >> 6, c = idx & 63;
    d[(size_t)(n0 + r) * IC + i0 + c] = t[r][c];
  }
}

// ---------------- 3) projections: out[o,n] = W[o,:]·X[:,n] + bias, bf16 out ----------------
__global__ void k_proj(const unsigned short* Wbf, const float* tb, const float* fb, const float* gb,
                       const unsigned short* XT, unsigned short* TH, unsigned short* PH,
                       unsigned short* GXT) {
  int which = blockIdx.z % 3, b = blockIdx.z / 3;
  const unsigned short* W = Wbf + (size_t)which * 65536;
  const float* bias = (which == 0) ? tb : (which == 1) ? fb : gb;
  unsigned short* out = ((which == 0) ? TH : (which == 1) ? PH : GXT) + (size_t)b * IC * NPOS;
  const unsigned short* xt = XT + (size_t)b * NPOS * IC;
  int obase = blockIdx.y * 16, nchunk = blockIdx.x * 256;
  int lane = threadIdx.x & 31, g = lane >> 4, m = lane & 15;
  bf16x16 a[8];
#pragma unroll
  for (int kk = 0; kk < 8; ++kk) a[kk] = load_a_frag(W, IC, obase, kk * 32);
  float bv[8];
#pragma unroll
  for (int r = 0; r < 8; ++r) bv[r] = bias[obase + r + 8 * g];
  for (int nt = 0; nt < 256; nt += 16) {
    f32x8 c = zero8();
#pragma unroll
    for (int kk = 0; kk < 8; ++kk)
      c = wmma_bf(a[kk], load_b_frag(xt, IC, nchunk + nt, kk * 32), c);
#pragma unroll
    for (int r = 0; r < 8; ++r)
      out[(size_t)(obase + r + 8 * g) * NPOS + nchunk + nt + m] = f2bf(c[r] + bv[r]);
  }
}

// ---------------- 4) bf16 transpose [256,4096] -> [4096,256] ----------------
__global__ void k_trbf(const unsigned short* S, unsigned short* D) {
  __shared__ unsigned short t[64][65];
  int b = blockIdx.z;
  const unsigned short* s = S + (size_t)b * IC * NPOS;
  unsigned short* d = D + (size_t)b * NPOS * IC;
  int n0 = blockIdx.x * 64, i0 = blockIdx.y * 64;
#pragma unroll
  for (int k = 0; k < 16; ++k) {
    int idx = threadIdx.x + k * 256;
    int r = idx >> 6, c = idx & 63;
    t[c][r] = s[(size_t)(i0 + r) * NPOS + n0 + c];
  }
  __syncthreads();
#pragma unroll
  for (int k = 0; k < 16; ++k) {
    int idx = threadIdx.x + k * 256;
    int r = idx >> 6, c = idx & 63;
    d[(size_t)(n0 + r) * IC + i0 + c] = t[r][c];
  }
}

// ---------------- 5) flash spatial attention ----------------
// Stage one 32-key step (PHT 32x256 + GXT 256x32 bf16, 32 KB) into LDS per block,
// async + double-buffered when the CDNA5 async-LDS builtins are available.
__device__ __forceinline__ void stage_tiles(const unsigned short* pht, const unsigned short* gxt,
                                            unsigned short* dPH, unsigned short* dGX,
                                            int m0, int tid) {
#ifdef HAVE_ASYNC_LDS
#pragma unroll
  for (int k = 0; k < 8; ++k) {
    int c = tid + k * 128;
    int r = c >> 5, ch = c & 31;   // PHT: 32 rows x 32 chunks of 8 bf16
    __builtin_amdgcn_global_load_async_to_lds_b128(
        (as1_v4i)(pht + (size_t)(m0 + r) * IC + ch * 8),
        (as3_v4i)(dPH + r * IC + ch * 8), 0, 0);
    int r2 = c >> 2, ch2 = c & 3;  // GXT: 256 rows x 4 chunks of 8 bf16
    __builtin_amdgcn_global_load_async_to_lds_b128(
        (as1_v4i)(gxt + (size_t)r2 * NPOS + m0 + ch2 * 8),
        (as3_v4i)(dGX + r2 * 32 + ch2 * 8), 0, 0);
  }
#else
#pragma unroll
  for (int k = 0; k < 8; ++k) {
    int c = tid + k * 128;
    int r = c >> 5, ch = c & 31;
    *(uint4*)(dPH + r * IC + ch * 8) = *(const uint4*)(pht + (size_t)(m0 + r) * IC + ch * 8);
    int r2 = c >> 2, ch2 = c & 3;
    *(uint4*)(dGX + r2 * 32 + ch2 * 8) = *(const uint4*)(gxt + (size_t)r2 * NPOS + m0 + ch2 * 8);
  }
#endif
}

#ifdef HAVE_ASYNC_LDS
#define NBUF 2
#else
#define NBUF 1
#endif

__global__ void __launch_bounds__(128) k_ys(const unsigned short* THT, const unsigned short* PHT,
                                            const unsigned short* GXT, float* YS) {
  __shared__ __align__(16) unsigned short sPH[NBUF][32 * IC];
  __shared__ __align__(16) unsigned short sGX[NBUF][IC * 32];
  __shared__ __align__(16) unsigned short pbuf[4][512];
  int b = blockIdx.y;
  int tid = threadIdx.x;
  int wave = tid >> 5, lane = tid & 31;
  int g = lane >> 4, m = lane & 15;
  int qbase = blockIdx.x * 64 + wave * 16;
  const unsigned short* tht = THT + (size_t)b * NPOS * IC;
  const unsigned short* pht = PHT + (size_t)b * NPOS * IC;
  const unsigned short* gxt = GXT + (size_t)b * IC * NPOS;
  unsigned short* pb = pbuf[wave];

  bf16x16 a[8];
#pragma unroll
  for (int kk = 0; kk < 8; ++kk) a[kk] = load_a_frag(tht, IC, qbase, kk * 32);

  f32x8 o[16];
#pragma unroll
  for (int t = 0; t < 16; ++t) o[t] = zero8();
  f32x8 lacc = zero8();

#ifdef HAVE_ASYNC_LDS
  stage_tiles(pht, gxt, sPH[0], sGX[0], 0, tid);
  __builtin_amdgcn_s_wait_asynccnt(0);
  __syncthreads();
#endif
  int cur = 0;
  for (int m0 = 0; m0 < NPOS; m0 += 32) {
#ifdef HAVE_ASYNC_LDS
    int nxt = cur ^ 1;
    if (m0 + 32 < NPOS)
      stage_tiles(pht, gxt, sPH[nxt], sGX[nxt], m0 + 32, tid);  // DMA overlaps WMMAs below
#else
    __syncthreads();
    stage_tiles(pht, gxt, sPH[0], sGX[0], m0, tid);
    __syncthreads();
#endif
    const unsigned short* cPH = sPH[cur];
    const unsigned short* cGX = sGX[cur];
    // S = A(16 queries x 256) @ PH(256 x 32 keys), fp32 accumulate
    f32x8 s0 = zero8(), s1 = zero8();
#pragma unroll
    for (int kk = 0; kk < 8; ++kk) {
      s0 = wmma_bf(a[kk], load_b_frag(cPH, IC, 0, kk * 32), s0);
      s1 = wmma_bf(a[kk], load_b_frag(cPH, IC, 16, kk * 32), s1);
    }
    // P = exp(S) (scores bounded; no max subtraction needed, exact after normalization)
    f32x8 e0, e1;
#pragma unroll
    for (int r = 0; r < 8; ++r) { e0[r] = __expf(s0[r]); e1[r] = __expf(s1[r]); }
    lacc += e0 + e1;
    // route P (C layout) through per-wave LDS into A layout (same-wave DS ops are in order)
#pragma unroll
    for (int r = 0; r < 8; ++r) {
      pb[(r + 8 * g) * 32 + m] = f2bf(e0[r]);
      pb[(r + 8 * g) * 32 + m + 16] = f2bf(e1[r]);
    }
    FragU pu;
    const unsigned short* pp = pb + m * 32 + 8 * g;
    pu.q[0] = *(const uint4*)pp;
    pu.q[1] = *(const uint4*)(pp + 16);
    bf16x16 pf = pu.v;
    // O += P(16x32) @ G(32x256): B^T rows come from staged GXT[o][m]
#pragma unroll
    for (int ot = 0; ot < 16; ++ot)
      o[ot] = wmma_bf(pf, load_b_frag(cGX, 32, ot * 16, 0), o[ot]);
#ifdef HAVE_ASYNC_LDS
    __builtin_amdgcn_s_wait_asynccnt(0);
    __syncthreads();
    cur = nxt;
#endif
  }
  // row sums: reduce across the 16 lanes of each half
  float inv[8];
#pragma unroll
  for (int r = 0; r < 8; ++r) {
    float s = lacc[r];
#pragma unroll
    for (int sh = 1; sh < 16; sh <<= 1) s += __shfl_xor(s, sh, 32);
    inv[r] = 1.0f / s;
  }
  float* ysb = YS + (size_t)b * NPOS * IC;
#pragma unroll
  for (int ot = 0; ot < 16; ++ot)
#pragma unroll
    for (int r = 0; r < 8; ++r)
      ysb[(size_t)(qbase + r + 8 * g) * IC + ot * 16 + m] = o[ot][r] * inv[r];
}

// ---------------- 6) channel attention scores: FC[o,d] = TH[o,:]·PH[d,:] over n ----------------
__global__ void k_fc(const unsigned short* TH, const unsigned short* PH, float* FC) {
  int b = blockIdx.z;
  int o0 = blockIdx.y * 16, d0 = blockIdx.x * 16;
  int lane = threadIdx.x & 31, g = lane >> 4, m = lane & 15;
  const unsigned short* th = TH + (size_t)b * IC * NPOS;
  const unsigned short* ph = PH + (size_t)b * IC * NPOS;
  f32x8 c = zero8();
  for (int k = 0; k < NPOS; k += 32)
    c = wmma_bf(load_a_frag(th, NPOS, o0, k), load_b_frag(ph, NPOS, d0, k), c);
#pragma unroll
  for (int r = 0; r < 8; ++r)
    FC[((size_t)b * IC + o0 + r + 8 * g) * IC + d0 + m] = c[r];
}

// ---------------- 7) softmax rows of FC, write transposed bf16 FCT[d][o] ----------------
__global__ void k_fcsm(const float* FC, unsigned short* FCT) {
  __shared__ float red[256];
  int b = blockIdx.y, orow = blockIdx.x, t = threadIdx.x;
  float e = __expf(FC[((size_t)b * IC + orow) * IC + t]);
  red[t] = e;
  __syncthreads();
#pragma unroll
  for (int s = 128; s > 0; s >>= 1) {
    if (t < s) red[t] += red[t + s];
    __syncthreads();
  }
  float inv = 1.0f / red[0];
  FCT[((size_t)b * IC + t) * IC + orow] = f2bf(e * inv);
}

// ---------------- 8) YC[n,d] = GX[n,:] @ softmax(FC) ----------------
__global__ void k_yc(const unsigned short* GX, const unsigned short* FCT, float* YC) {
  int b = blockIdx.y, n0 = blockIdx.x * 16;
  int lane = threadIdx.x & 31, g = lane >> 4, m = lane & 15;
  const unsigned short* gx = GX + (size_t)b * NPOS * IC;
  const unsigned short* fct = FCT + (size_t)b * IC * IC;
  bf16x16 a[8];
#pragma unroll
  for (int kk = 0; kk < 8; ++kk) a[kk] = load_a_frag(gx, IC, n0, kk * 32);
  float* ycb = YC + (size_t)b * NPOS * IC;
  for (int dt = 0; dt < 16; ++dt) {
    f32x8 c = zero8();
#pragma unroll
    for (int kk = 0; kk < 8; ++kk)
      c = wmma_bf(a[kk], load_b_frag(fct, IC, dt * 16, kk * 32), c);
#pragma unroll
    for (int r = 0; r < 8; ++r)
      ycb[(size_t)(n0 + r + 8 * g) * IC + dt * 16 + m] = c[r];
  }
}

// ---------------- 9) z = x + gs*(Ws@ys + Wsb) + gc*(Wc@yc + Wcb) ----------------
__global__ void k_z(const float* x, const float* YS, const float* YC,
                    const float* Wsw, const float* Wsb, const float* Wcw, const float* Wcb,
                    const float* gs_, const float* gc_, float* Z) {
  __shared__ float sWs[4096], sWc[4096], sYs[4096], sYc[4096];
  int b = blockIdx.y, p0 = blockIdx.x * 64, tid = threadIdx.x;
  int k0 = p0 >> 12;       // y_s channel-fold index
  int nbS = p0 & 4095;     // y_s position base
  int rowC = p0 >> 8;      // y_c row fold
  int colC = p0 & 255;     // y_c column base
#pragma unroll
  for (int k = 0; k < 16; ++k) {
    int idx = tid + k * 256;
    sWs[idx] = Wsw[idx];
    sWc[idx] = Wcw[idx];
    int ci = idx >> 6, t = idx & 63;
    sYs[idx] = YS[((size_t)b * NPOS + nbS + t) * IC + ci * 4 + k0];
    sYc[idx] = YC[((size_t)b * NPOS + ci * 64 + rowC) * IC + colC + t];
  }
  __syncthreads();
  float gs = gs_[0], gc = gc_[0];
#pragma unroll
  for (int k = 0; k < 16; ++k) {
    int idx = tid + k * 256;
    int co = idx >> 6, t = idx & 63;
    float as = Wsb[co], ac = Wcb[co];
    for (int ci = 0; ci < 64; ++ci) {
      as += sWs[co * 64 + ci] * sYs[ci * 64 + t];
      ac += sWc[co * 64 + ci] * sYc[ci * 64 + t];
    }
    size_t zi = ((size_t)b * 64 + co) * HW + p0 + t;
    Z[zi] = x[zi] + gs * as + gc * ac;
  }
}

// ---------------- 10) conv 3x3, 64->16, relu ----------------
__global__ void k_conv1(const float* Z, const float* w, const float* bias, float* Hm) {
  __shared__ float tile[18][18];
  __shared__ float wsm[144];
  int b = blockIdx.z, y0 = blockIdx.y * 16, x0 = blockIdx.x * 16;
  int ty = threadIdx.x >> 4, tx = threadIdx.x & 15;
  float acc[16];
#pragma unroll
  for (int oc = 0; oc < 16; ++oc) acc[oc] = bias[oc];
  for (int ic = 0; ic < 64; ++ic) {
    __syncthreads();
    for (int idx = threadIdx.x; idx < 324; idx += 256) {
      int r = idx / 18, c2 = idx % 18;
      int yy = y0 + r - 1, xx = x0 + c2 - 1;
      tile[r][c2] = (yy >= 0 && yy < 128 && xx >= 0 && xx < 128)
                        ? Z[((size_t)b * 64 + ic) * HW + yy * 128 + xx]
                        : 0.f;
    }
    if (threadIdx.x < 144)
      wsm[threadIdx.x] = w[(threadIdx.x / 9) * 576 + ic * 9 + threadIdx.x % 9];
    __syncthreads();
#pragma unroll
    for (int oc = 0; oc < 16; ++oc) {
      float a = acc[oc];
#pragma unroll
      for (int ky = 0; ky < 3; ++ky)
#pragma unroll
        for (int kx = 0; kx < 3; ++kx)
          a += wsm[oc * 9 + ky * 3 + kx] * tile[ty + ky][tx + kx];
      acc[oc] = a;
    }
  }
#pragma unroll
  for (int oc = 0; oc < 16; ++oc)
    Hm[((size_t)b * 16 + oc) * HW + (y0 + ty) * 128 + x0 + tx] = fmaxf(acc[oc], 0.f);
}

// ---------------- 11) conv 3x3, 16->1 ----------------
__global__ void k_conv2(const float* Hm, const float* w, const float* bias, float* Out) {
  __shared__ float tile[18][18];
  int b = blockIdx.z, y0 = blockIdx.y * 16, x0 = blockIdx.x * 16;
  int ty = threadIdx.x >> 4, tx = threadIdx.x & 15;
  float acc = bias[0];
  for (int ic = 0; ic < 16; ++ic) {
    __syncthreads();
    for (int idx = threadIdx.x; idx < 324; idx += 256) {
      int r = idx / 18, c2 = idx % 18;
      int yy = y0 + r - 1, xx = x0 + c2 - 1;
      tile[r][c2] = (yy >= 0 && yy < 128 && xx >= 0 && xx < 128)
                        ? Hm[((size_t)b * 16 + ic) * HW + yy * 128 + xx]
                        : 0.f;
    }
    __syncthreads();
#pragma unroll
    for (int ky = 0; ky < 3; ++ky)
#pragma unroll
      for (int kx = 0; kx < 3; ++kx)
        acc += w[ic * 9 + ky * 3 + kx] * tile[ty + ky][tx + kx];
  }
  Out[(size_t)b * HW + (y0 + ty) * 128 + x0 + tx] = acc;
}

extern "C" void kernel_launch(void* const* d_in, const int* in_sizes, int n_in,
                              void* d_out, int out_size, void* d_ws, size_t ws_size,
                              hipStream_t stream) {
  const float* x = (const float*)d_in[0];
  const float* g_w = (const float*)d_in[1];
  const float* g_b = (const float*)d_in[2];
  const float* th_w = (const float*)d_in[3];
  const float* th_b = (const float*)d_in[4];
  const float* ph_w = (const float*)d_in[5];
  const float* ph_b = (const float*)d_in[6];
  const float* Ws_w = (const float*)d_in[7];
  const float* Ws_b = (const float*)d_in[8];
  const float* Wc_w = (const float*)d_in[9];
  const float* Wc_b = (const float*)d_in[10];
  const float* gam_s = (const float*)d_in[11];
  const float* gam_c = (const float*)d_in[12];
  const float* m1w = (const float*)d_in[13];
  const float* m1b = (const float*)d_in[14];
  const float* m2w = (const float*)d_in[15];
  const float* m2b = (const float*)d_in[16];

  float* out = (float*)d_out;
  float* logit = out;              // [4,1,128,128]
  float* Z = out + 4 * HW;         // [4,64,128,128]

  char* ws = (char*)d_ws;
  size_t off = 0;
  auto alloc = [&](size_t bytes) {
    size_t o = off;
    off += (bytes + 255) & ~(size_t)255;
    return o;
  };
  const size_t BF_MAT = (size_t)4 * IC * NPOS * 2;  // 8 MB
  unsigned short* XT = (unsigned short*)(ws + alloc(BF_MAT));
  unsigned short* TH = (unsigned short*)(ws + alloc(BF_MAT));
  unsigned short* PH = (unsigned short*)(ws + alloc(BF_MAT));
  unsigned short* GXT = (unsigned short*)(ws + alloc(BF_MAT));
  unsigned short* THT = (unsigned short*)(ws + alloc(BF_MAT));
  unsigned short* PHT = (unsigned short*)(ws + alloc(BF_MAT));
  unsigned short* GX = (unsigned short*)(ws + alloc(BF_MAT));
  unsigned short* Wbf = (unsigned short*)(ws + alloc((size_t)3 * 65536 * 2));
  float* YS = (float*)(ws + alloc((size_t)4 * NPOS * IC * 4));
  float* YC = (float*)(ws + alloc((size_t)4 * NPOS * IC * 4));
  float* FC = (float*)(ws + alloc((size_t)4 * IC * IC * 4));
  unsigned short* FCT = (unsigned short*)(ws + alloc((size_t)4 * IC * IC * 2));
  float* HMID = (float*)(ws + alloc((size_t)4 * 16 * HW * 4));
  (void)ws_size; (void)in_sizes; (void)n_in; (void)out_size;

  // 1) weights -> bf16 (order: theta, phi, g)
  k_w2bf<<<dim3(768), dim3(256), 0, stream>>>(th_w, ph_w, g_w, Wbf);
  // 2) x -> XT bf16
  k_xT<<<dim3(64, 4, 4), dim3(256), 0, stream>>>(x, XT);
  // 3) projections
  k_proj<<<dim3(16, 16, 12), dim3(32), 0, stream>>>(Wbf, th_b, ph_b, g_b, XT, TH, PH, GXT);
  // 4) transposes: TH->THT, PH->PHT, GXT->GX
  k_trbf<<<dim3(64, 4, 4), dim3(256), 0, stream>>>(TH, THT);
  k_trbf<<<dim3(64, 4, 4), dim3(256), 0, stream>>>(PH, PHT);
  k_trbf<<<dim3(64, 4, 4), dim3(256), 0, stream>>>(GXT, GX);
  // 5) flash spatial attention -> YS
  k_ys<<<dim3(64, 4), dim3(128), 0, stream>>>(THT, PHT, GXT, YS);
  // 6) channel scores -> FC
  k_fc<<<dim3(16, 16, 4), dim3(32), 0, stream>>>(TH, PH, FC);
  // 7) softmax -> FCT (transposed bf16)
  k_fcsm<<<dim3(256, 4), dim3(256), 0, stream>>>(FC, FCT);
  // 8) channel attention apply -> YC
  k_yc<<<dim3(256, 4), dim3(32), 0, stream>>>(GX, FCT, YC);
  // 9) z
  k_z<<<dim3(256, 4), dim3(256), 0, stream>>>(x, YS, YC, Ws_w, Ws_b, Wc_w, Wc_b, gam_s, gam_c, Z);
  // 10) conv1 + relu
  k_conv1<<<dim3(8, 8, 4), dim3(256), 0, stream>>>(Z, m1w, m1b, HMID);
  // 11) conv2 -> logit
  k_conv2<<<dim3(8, 8, 4), dim3(256), 0, stream>>>(HMID, m2w, m2b, logit);
}
